// Ac4kLinearOp_51505247814404
// MI455X (gfx1250) — matrix operand order repaired and checked by reading, not compiled, not run on hardware
//
#include <hip/hip_runtime.h>
#include <hip/hip_bf16.h>
#include <hip/hip_fp16.h>

// ---------------------------------------------------------------------------
// NVFP4 block-scaled GEMM for MI455X (gfx1250).
// out[m,n] = sum_k fp4(x[m,k])*xs[m,k/16]*xgs * fp4(w[n,k])*ws[n,k/16]*wgs + bias[n]
// Memory-bound (~275MB @ 23.3TB/s ~ 12us vs <1us of WMMA math), so:
//   * stream weight codes exactly once, 64B-contiguous per lane
//   * pre-dequant x to f16 in d_ws (1MB, L2-resident, reused by all WGs)
//   * 3-op/element FP4 decode via v_perm_b32 + v_pk_mul_f16
//   * v_wmma_f32_16x16x32_f16 for the math
// ---------------------------------------------------------------------------

typedef _Float16 v16h __attribute__((ext_vector_type(16)));
typedef _Float16 h2   __attribute__((ext_vector_type(2)));
typedef float    v8f  __attribute__((ext_vector_type(8)));

union H2U   { unsigned int u; h2 h; };
union FragA { unsigned int u[8]; v16h v; };
union FragB { h2 h[8]; v16h v; };

// Decode 4 FP4 codes (one per uint4 component, values 0..15) into 2 packed-f16
// pairs, scaled by sv.  Magnitude high-byte LUT via v_perm_b32:
//   mags 0..3 -> bytes of 0x3E3C3800, mags 4..7 -> bytes of 0x46444240.
__device__ __forceinline__ void decode4(uint4 q, h2 sv, h2* dst) {
  unsigned int V   = (unsigned int)q.x | ((unsigned int)q.y << 8) |
                     ((unsigned int)q.z << 16) | ((unsigned int)q.w << 24);
  unsigned int sel = V & 0x07070707u;          // 3-bit magnitude per byte
  unsigned int sgn = (V & 0x08080808u) << 4;   // sign bit -> byte bit7
  unsigned int hb  = __builtin_amdgcn_perm(0x46444240u, 0x3E3C3800u, sel) | sgn;
  H2U p0, p1;
  // expand high-byte vector [h3 h2 h1 h0] -> packed f16 (h<<8 per element)
  p0.u = __builtin_amdgcn_perm(0u, hb, 0x01040004u);  // [h1,0,h0,0]
  p1.u = __builtin_amdgcn_perm(0u, hb, 0x03040204u);  // [h3,0,h2,0]
  dst[0] = p0.h * sv;   // v_pk_mul_f16
  dst[1] = p1.h * sv;
}

// ---- pre-kernel: dequantize x (codes*block_scale*global_scale) to f16 ------
__global__ __launch_bounds__(256)
void dequant_x_kernel(const int* __restrict__ x, const float* __restrict__ xs,
                      const float* __restrict__ xgs, _Float16* __restrict__ out,
                      int total) {
  int idx = blockIdx.x * blockDim.x + threadIdx.x;
  if (idx >= total) return;
  unsigned int c   = (unsigned int)x[idx] & 15u;
  unsigned int mag = c & 7u;
  // f32 bits of E2M1 magnitude: mag>=2 -> 0x3F000000 + (mag<<22); 1->0.5; 0->0
  unsigned int bits = (mag < 2u) ? (mag ? 0x3F000000u : 0u)
                                 : (0x3F000000u + (mag << 22));
  bits |= (c & 8u) << 28;
  float v = __uint_as_float(bits) * xs[idx >> 4] * xgs[0];
  out[idx] = (_Float16)v;
}

// ---- main kernel: 4 waves/WG, each wave = one 16-col N tile x four M tiles -
__global__ __launch_bounds__(128)
void nvfp4_gemm_kernel(const int* __restrict__ w,
                       const float* __restrict__ wscale,
                       const float* __restrict__ wgscale,
                       const _Float16* __restrict__ xf16,
                       const float* __restrict__ bias,
                       __hip_bfloat16* __restrict__ out,
                       int N, int K) {
  const int lane  = threadIdx.x & 31;
  const int wave  = threadIdx.x >> 5;
  const int n0    = blockIdx.x * 64 + wave * 16;
  const int n     = n0 + (lane & 15);
  const int khalf = (lane & 16);        // B frag: lanes 16-31 hold K+16..K+31
  const int ak    = (lane & 16);        // A frag: lanes 16-31 hold K+16..K+31
  const int arow  = (lane & 15);
  const int kb16  = K >> 4;

  const float wgs = wgscale[0];
  const int* wrow = w + (size_t)n * K;

  v8f acc[4] = {v8f{}, v8f{}, v8f{}, v8f{}};

  for (int k0 = 0; k0 < K; k0 += 32) {
    const int kb = k0 + khalf;
    __builtin_prefetch((const void*)(wrow + kb + 128), 0, 1);  // global_prefetch

    // one lane = one contiguous 64B run of 16 codes = exactly one scale block
    const uint4* qp = (const uint4*)(wrow + kb);
    uint4 q0 = qp[0], q1 = qp[1], q2 = qp[2], q3 = qp[3];

    float bs = wscale[n * kb16 + (kb >> 4)] * wgs;
    _Float16 hs = (_Float16)bs;
    h2 sv; sv[0] = hs; sv[1] = hs;

    FragB fb;
    decode4(q0, sv, &fb.h[0]);
    decode4(q1, sv, &fb.h[2]);
    decode4(q2, sv, &fb.h[4]);
    decode4(q3, sv, &fb.h[6]);

#pragma unroll
    for (int mt = 0; mt < 4; ++mt) {
      const uint4* ap =
          (const uint4*)(xf16 + (size_t)((mt * 16 + arow) * K) + k0 + ak);
      uint4 lo = ap[0];   // K offsets ak .. ak+7
      uint4 hi = ap[1];   // K offsets ak+8 .. ak+15
      FragA fa;
      fa.u[0] = lo.x; fa.u[1] = lo.y; fa.u[2] = lo.z; fa.u[3] = lo.w;
      fa.u[4] = hi.x; fa.u[5] = hi.y; fa.u[6] = hi.z; fa.u[7] = hi.w;
      acc[mt] = __builtin_amdgcn_wmma_f32_16x16x32_f16(
          false, fa.v, false, fb.v, (short)0, acc[mt], false, false);
    }
  }

  // epilogue: C/D layout -> VGPR j is row j (lanes 0-15) / row j+8 (16-31)
  const float bn   = bias[n];
  const int  rbase = (lane & 16) >> 1;
#pragma unroll
  for (int mt = 0; mt < 4; ++mt) {
#pragma unroll
    for (int j = 0; j < 8; ++j) {
      int row = mt * 16 + rbase + j;
      out[(size_t)row * N + n] = __float2bfloat16(acc[mt][j] + bn);
    }
  }
}

extern "C" void kernel_launch(void* const* d_in, const int* in_sizes, int n_in,
                              void* d_out, int out_size, void* d_ws, size_t ws_size,
                              hipStream_t stream) {
  const int*   x    = (const int*)d_in[0];
  const float* xs   = (const float*)d_in[1];
  const float* xgs  = (const float*)d_in[2];
  const int*   w    = (const int*)d_in[3];
  const float* wsc  = (const float*)d_in[4];
  const float* wgs  = (const float*)d_in[5];
  const float* bias = (const float*)d_in[6];
  __hip_bfloat16* out = (__hip_bfloat16*)d_out;

  const int N = in_sizes[6];            // bias length
  const int K = in_sizes[3] / N;        // weight is N*K
  const int M = in_sizes[0] / K;        // 64

  _Float16* xf16 = (_Float16*)d_ws;     // M*K f16 = 1 MB, L2-resident
  const int totalx = M * K;
  dequant_x_kernel<<<(totalx + 255) / 256, 256, 0, stream>>>(x, xs, xgs, xf16,
                                                             totalx);
  nvfp4_gemm_kernel<<<N / 64, 128, 0, stream>>>(w, wsc, wgs, xf16, bias, out,
                                                N, K);
}
